// Graphgps_4320737100495
// MI455X (gfx1250) — matrix operand order repaired and checked by reading, not compiled, not run on hardware
//
#include <hip/hip_runtime.h>
#include <hip/hip_bf16.h>
#include <stdint.h>

// ---------------------------------------------------------------------------
// GraphGPS forward for MI455X (gfx1250): bf16 WMMA GEMMs + flash attention.
//  - GEMM: 1x4 register blocking (16x64 strip/wave), A-fragment reuse.
//  - Flash attention: 64 keys/iteration, amortized softmax reductions.
// ---------------------------------------------------------------------------

#define NN   4096     // nodes
#define CC   128      // channels
#define HH   4        // heads
#define DD   32       // head dim
#define GG   64       // graphs
#define NCLS 10
#define LL   2

typedef __attribute__((ext_vector_type(16))) __bf16 v16bf;
typedef __attribute__((ext_vector_type(8)))  float  v8f;

union FragU { v16bf v; uint4 q[2]; };

// A fragment (16x32 bf16, row-major source, row = lane%16):
// lanes 0-15 take K=[0..7]+[16..23], lanes 16-31 take K=[8..15]+[24..31]
__device__ __forceinline__ v16bf load_fragA(const __bf16* base, int stride, int lane) {
    int m  = lane & 15;
    int hf = (lane >> 4) & 1;
    const __bf16* p = base + (size_t)m * stride + hf * 8;
    FragU f;
    f.q[0] = *(const uint4*)(p);
    f.q[1] = *(const uint4*)(p + 16);
    return f.v;
}

// B fragment (32x16 bf16, source stores each COLUMN contiguously over K,
// i.e. rows of W / rows of K-matrix / rows of V^T). Column = lane%16,
// lanes 0-15 take K=[0..7]+[16..23], lanes 16-31 take K=[8..15]+[24..31].
__device__ __forceinline__ v16bf load_fragB(const __bf16* base, int stride, int lane) {
    int n  = lane & 15;
    int hf = (lane >> 4) & 1;
    const __bf16* p = base + (size_t)n * stride + hf * 8;
    FragU f;
    f.q[0] = *(const uint4*)(p);
    f.q[1] = *(const uint4*)(p + 16);
    return f.v;
}

__device__ __forceinline__ v8f wmma_bf16(v16bf a, v16bf b, v8f c) {
    return __builtin_amdgcn_wmma_f32_16x16x32_bf16(
        false, a, false, b, (short)0, c, false, false);
}

// ---------------------------------------------------------------------------
// GEMM: out[M x N] = act(A[M x K](bf16) @ W[N x K]^T(bf16) + bias) (+resid)
// One wave computes a 16x64 output strip (4 accumulators, A reused 4x).
// Requires M%16==0, N%64==0, K%32==0.
// ---------------------------------------------------------------------------
__global__ void __launch_bounds__(256)
gemm_bf16_k(const __bf16* __restrict__ A, const __bf16* __restrict__ W,
            const float* __restrict__ bias, const float* __restrict__ resid,
            float* __restrict__ outF, __bf16* __restrict__ outB,
            int M, int N, int K, int relu)
{
    int lane   = threadIdx.x;
    int strip  = blockIdx.x * blockDim.y + threadIdx.y;
    int stripsN = N >> 6;                     // 64-wide strips
    int total   = (M >> 4) * stripsN;
    if (strip >= total) return;
    int tm = strip / stripsN;
    int tg = strip - tm * stripsN;

    const __bf16* Ab = A + (size_t)tm * 16 * K;
    const __bf16* Wb = W + (size_t)(tg * 64) * K;

    v8f c0 = {}, c1 = {}, c2 = {}, c3 = {};
    for (int k = 0; k < K; k += 32) {
        v16bf a  = load_fragA(Ab + k, K, lane);
        v16bf b0 = load_fragB(Wb + k, K, lane);
        v16bf b1 = load_fragB(Wb + (size_t)16 * K + k, K, lane);
        v16bf b2 = load_fragB(Wb + (size_t)32 * K + k, K, lane);
        v16bf b3 = load_fragB(Wb + (size_t)48 * K + k, K, lane);
        c0 = wmma_bf16(a, b0, c0);
        c1 = wmma_bf16(a, b1, c1);
        c2 = wmma_bf16(a, b2, c2);
        c3 = wmma_bf16(a, b3, c3);
    }

    int hf = (lane >> 4) & 1, col = lane & 15;
    v8f* acc[4] = {&c0, &c1, &c2, &c3};
#pragma unroll
    for (int j = 0; j < 4; j++) {
        int gcol = tg * 64 + j * 16 + col;
        float bv = bias ? bias[gcol] : 0.0f;
#pragma unroll
        for (int i = 0; i < 8; i++) {
            int grow = tm * 16 + i + 8 * hf;
            float v = (*acc[j])[i] + bv;
            if (relu) v = fmaxf(v, 0.0f);
            size_t o = (size_t)grow * N + gcol;
            if (resid) v += resid[o];
            if (outF) outF[o] = v;
            if (outB) outB[o] = (__bf16)v;
        }
    }
}

// ---------------------------------------------------------------------------
// Flash attention: one wave per (16-query tile, head); streams 64 keys/iter.
// qkv: [N x 384] bf16 (q|k|v, bias added). vt: [128 x N] bf16 (V^T).
// ao:  [N x 128] bf16 per-head concat output (pre out-projection).
// ---------------------------------------------------------------------------
__global__ void __launch_bounds__(128)
flash_attn_k(const __bf16* __restrict__ qkv, const __bf16* __restrict__ vt,
             __bf16* __restrict__ ao)
{
    __shared__ __bf16 plds[4][16 * 64];
    int lane = threadIdx.x;
    int head = threadIdx.y;         // blockDim.y == 4
    int qt   = blockIdx.x;          // 0..255
    int hf = (lane >> 4) & 1, col = lane & 15;
    const float scale = 0.17677669529663687f;   // 1/sqrt(32)

    v16bf aq = load_fragA(qkv + (size_t)(qt * 16) * 384 + head * DD, 384, lane);

    float mi[8], li[8];
    v8f o0 = {}, o1 = {};
#pragma unroll
    for (int i = 0; i < 8; i++) { mi[i] = -1e30f; li[i] = 0.0f; }

    __bf16* pl = plds[head];
    v8f z = {};

    for (int kb = 0; kb < NN; kb += 64) {
        const __bf16* kbase = qkv + (size_t)kb * 384 + CC + head * DD;
        v16bf bk0 = load_fragB(kbase, 384, lane);
        v16bf bk1 = load_fragB(kbase + (size_t)16 * 384, 384, lane);
        v16bf bk2 = load_fragB(kbase + (size_t)32 * 384, 384, lane);
        v16bf bk3 = load_fragB(kbase + (size_t)48 * 384, 384, lane);
        v8f s0 = wmma_bf16(aq, bk0, z);
        v8f s1 = wmma_bf16(aq, bk1, z);
        v8f s2 = wmma_bf16(aq, bk2, z);
        v8f s3 = wmma_bf16(aq, bk3, z);

#pragma unroll
        for (int i = 0; i < 8; i++) {
            float a0 = s0[i] * scale, a1 = s1[i] * scale;
            float a2 = s2[i] * scale, a3 = s3[i] * scale;
            float mx = fmaxf(fmaxf(a0, a1), fmaxf(a2, a3));
            for (int msk = 1; msk < 16; msk <<= 1)
                mx = fmaxf(mx, __shfl_xor(mx, msk, 32));
            float mn    = fmaxf(mi[i], mx);
            float alpha = __expf(mi[i] - mn);
            float p0 = __expf(a0 - mn), p1 = __expf(a1 - mn);
            float p2 = __expf(a2 - mn), p3 = __expf(a3 - mn);
            float ps = (p0 + p1) + (p2 + p3);
            for (int msk = 1; msk < 16; msk <<= 1)
                ps += __shfl_xor(ps, msk, 32);
            li[i] = li[i] * alpha + ps;
            mi[i] = mn;
            o0[i] *= alpha;
            o1[i] *= alpha;
            int r = i + 8 * hf;
            pl[r * 64 + col]      = (__bf16)p0;
            pl[r * 64 + 16 + col] = (__bf16)p1;
            pl[r * 64 + 32 + col] = (__bf16)p2;
            pl[r * 64 + 48 + col] = (__bf16)p3;
        }
        asm volatile("s_wait_dscnt 0" ::: "memory");

        v16bf ap0 = load_fragA(pl, 64, lane);        // keys kb+0..31
        v16bf ap1 = load_fragA(pl + 32, 64, lane);   // keys kb+32..63
        const __bf16* vb0 = vt + (size_t)(head * DD) * NN + kb;
        const __bf16* vb1 = vt + (size_t)(head * DD + 16) * NN + kb;
        v16bf bv00 = load_fragB(vb0, NN, lane);        // dims 0-15,  keys 0-31
        v16bf bv01 = load_fragB(vb0 + 32, NN, lane);   // dims 0-15,  keys 32-63
        v16bf bv10 = load_fragB(vb1, NN, lane);        // dims 16-31, keys 0-31
        v16bf bv11 = load_fragB(vb1 + 32, NN, lane);   // dims 16-31, keys 32-63
        o0 = wmma_bf16(ap0, bv00, o0);
        o0 = wmma_bf16(ap1, bv01, o0);
        o1 = wmma_bf16(ap0, bv10, o1);
        o1 = wmma_bf16(ap1, bv11, o1);
    }

#pragma unroll
    for (int i = 0; i < 8; i++) {
        float inv = 1.0f / li[i];
        int grow = qt * 16 + i + 8 * hf;
        ao[(size_t)grow * CC + head * DD + col]      = (__bf16)(o0[i] * inv);
        ao[(size_t)grow * CC + head * DD + 16 + col] = (__bf16)(o1[i] * inv);
    }
}

// ---------------------------------------------------------------------------
// BatchNorm1d (training-mode biased stats) over rows, per channel.
// ---------------------------------------------------------------------------
__global__ void __launch_bounds__(256)
bn_k(const float* __restrict__ in, const float* __restrict__ g,
     const float* __restrict__ b, float* __restrict__ outF,
     __bf16* __restrict__ outB, int M, int C)
{
    __shared__ float s1[256], s2[256];
    int c = blockIdx.x, t = threadIdx.x;
    float sum = 0.f, sq = 0.f;
    for (int r = t; r < M; r += 256) {
        float v = in[(size_t)r * C + c];
        sum += v; sq += v * v;
    }
    s1[t] = sum; s2[t] = sq;
    __syncthreads();
    for (int o = 128; o > 0; o >>= 1) {
        if (t < o) { s1[t] += s1[t + o]; s2[t] += s2[t + o]; }
        __syncthreads();
    }
    float mu  = s1[0] / M;
    float var = s2[0] / M - mu * mu;
    float is  = rsqrtf(var + 1e-5f) * g[c];
    float bb  = b[c];
    for (int r = t; r < M; r += 256) {
        size_t o = (size_t)r * C + c;
        float v = (in[o] - mu) * is + bb;
        if (outF) outF[o] = v;
        if (outB) outB[o] = (__bf16)v;
    }
}

// ---------------------------------------------------------------------------
// Small elementwise / scatter kernels
// ---------------------------------------------------------------------------
__global__ void cast_f32_bf16_k(const float* __restrict__ in,
                                __bf16* __restrict__ out, int n) {
    int i = blockIdx.x * blockDim.x + threadIdx.x;
    if (i < n) out[i] = (__bf16)in[i];
}

__global__ void fill_f32_k(float* p, float v, int n) {
    int i = blockIdx.x * blockDim.x + threadIdx.x;
    if (i < n) p[i] = v;
}

__global__ void deg_edge_k(const int* __restrict__ dst, float* deg, int E) {
    int i = blockIdx.x * blockDim.x + threadIdx.x;
    if (i < E) atomicAdd(&deg[dst[i]], 1.0f);
}

__global__ void dinv_k(const float* __restrict__ deg, float* dinv, int n) {
    int i = blockIdx.x * blockDim.x + threadIdx.x;
    if (i < n) dinv[i] = (deg[i] > 0.f) ? rsqrtf(deg[i]) : 0.f;
}

// loc = xw * dinv^2 (self loop) + h (residual)
__global__ void gcn_self_k(const float* __restrict__ xw,
                           const float* __restrict__ dinv,
                           const float* __restrict__ h,
                           float* __restrict__ loc, int n) {
    int i = blockIdx.x * blockDim.x + threadIdx.x;
    if (i >= n) return;
    int node = i >> 7;
    float d = dinv[node];
    loc[i] = xw[i] * d * d + h[i];
}

// loc[dst] += xw[src] * dinv[src]*dinv[dst] : one wave per edge, 4 ch/lane
__global__ void gcn_scatter_k(const float* __restrict__ xw,
                              const float* __restrict__ dinv,
                              const int* __restrict__ src,
                              const int* __restrict__ dst,
                              float* __restrict__ loc, int E) {
    int idx = blockIdx.x * blockDim.x + threadIdx.x;
    int e = idx >> 5;
    if (e >= E) return;
    int ln = idx & 31;
    int s = src[e], d = dst[e];
    float w = dinv[s] * dinv[d];
    const float* xs = xw + (size_t)s * CC + ln * 4;
    float* ld = loc + (size_t)d * CC + ln * 4;
#pragma unroll
    for (int j = 0; j < 4; j++) atomicAdd(ld + j, xs[j] * w);
}

__global__ void add2_k(const float* __restrict__ a, const float* __restrict__ b,
                       float* __restrict__ oF, __bf16* __restrict__ oB, int n) {
    int i = blockIdx.x * blockDim.x + threadIdx.x;
    if (i >= n) return;
    float v = a[i] + b[i];
    oF[i] = v;
    oB[i] = (__bf16)v;
}

__global__ void vtrans_k(const __bf16* __restrict__ qkv, __bf16* __restrict__ vt, int n) {
    int i = blockIdx.x * blockDim.x + threadIdx.x;
    if (i >= n) return;
    int node = i >> 7, c = i & 127;
    vt[(size_t)c * NN + node] = qkv[(size_t)node * 384 + 256 + c];
}

__global__ void lin2_k(const float* __restrict__ h, const float* __restrict__ w,
                       const float* __restrict__ b, float* __restrict__ logits) {
    int idx = blockIdx.x * blockDim.x + threadIdx.x;
    if (idx >= NN * NCLS) return;
    int n = idx / NCLS, j = idx - n * NCLS;
    const float* hr = h + (size_t)n * CC;
    const float* wr = w + (size_t)j * CC;
    float s = 0.f;
    for (int k = 0; k < CC; k++) s += hr[k] * wr[k];
    logits[idx] = s + b[j];
}

__global__ void pool_k(const float* __restrict__ logits,
                       const int* __restrict__ batch, float* pooled) {
    int idx = blockIdx.x * blockDim.x + threadIdx.x;
    if (idx >= NN * NCLS) return;
    int n = idx / NCLS, j = idx - n * NCLS;
    atomicAdd(&pooled[batch[n] * NCLS + j], logits[idx]);
}

__global__ void lsm_k(const float* __restrict__ pooled, float* __restrict__ out) {
    int g = threadIdx.x;
    if (g >= GG) return;
    const float* p = pooled + g * NCLS;
    float mx = -1e30f;
    for (int j = 0; j < NCLS; j++) mx = fmaxf(mx, p[j]);
    float s = 0.f;
    for (int j = 0; j < NCLS; j++) s += __expf(p[j] - mx);
    float lse = mx + __logf(s);
    for (int j = 0; j < NCLS; j++) out[g * NCLS + j] = p[j] - lse;
}

// ---------------------------------------------------------------------------
// Host launcher
// ---------------------------------------------------------------------------
static inline void* bump(char*& p, size_t bytes) {
    void* r = (void*)p;
    p += (bytes + 255) & ~(size_t)255;
    return r;
}

static inline void cast_w(const float* in, __bf16* out, int n, hipStream_t s) {
    cast_f32_bf16_k<<<(n + 255) / 256, 256, 0, s>>>(in, out, n);
}

static inline void launch_gemm(const __bf16* A, const __bf16* W, const float* bias,
                               const float* resid, float* outF, __bf16* outB,
                               int M, int N, int K, int relu, hipStream_t s) {
    int strips = (M / 16) * (N / 64);
    dim3 blk(32, 8);
    gemm_bf16_k<<<(strips + 7) / 8, blk, 0, s>>>(A, W, bias, resid, outF, outB,
                                                 M, N, K, relu);
}

extern "C" void kernel_launch(void* const* d_in, const int* in_sizes, int n_in,
                              void* d_out, int out_size, void* d_ws, size_t ws_size,
                              hipStream_t stream) {
    const float* x        = (const float*)d_in[0];
    const float* lin1_w   = (const float*)d_in[1];
    const float* lin1_b   = (const float*)d_in[2];
    const float* lin2_w   = (const float*)d_in[3];
    const float* lin2_b   = (const float*)d_in[4];
    const float* gcn_w    = (const float*)d_in[5];
    const float* in_w     = (const float*)d_in[6];
    const float* in_b     = (const float*)d_in[7];
    const float* out_w    = (const float*)d_in[8];
    const float* out_b    = (const float*)d_in[9];
    const float* w1       = (const float*)d_in[10];
    const float* b1       = (const float*)d_in[11];
    const float* w2       = (const float*)d_in[12];
    const float* b2       = (const float*)d_in[13];
    const float* n1g      = (const float*)d_in[14];
    const float* n1b      = (const float*)d_in[15];
    const float* n2g      = (const float*)d_in[16];
    const float* n2b      = (const float*)d_in[17];
    const float* n3g      = (const float*)d_in[18];
    const float* n3b      = (const float*)d_in[19];
    const int*   ei       = (const int*)d_in[20];
    const int*   batch    = (const int*)d_in[21];
    const int E = in_sizes[20] / 2;
    const int* src = ei;
    const int* dst = ei + E;

    char* p = (char*)d_ws;
    const int NC = NN * CC;
    __bf16* wl1_b = (__bf16*)bump(p, (size_t)CC * CC * 2);
    __bf16* wg_b  = (__bf16*)bump(p, (size_t)LL * CC * CC * 2);
    __bf16* wi_b  = (__bf16*)bump(p, (size_t)LL * 3 * CC * CC * 2);
    __bf16* wo_b  = (__bf16*)bump(p, (size_t)LL * CC * CC * 2);
    __bf16* w1_b  = (__bf16*)bump(p, (size_t)LL * 2 * CC * CC * 2);
    __bf16* w2_b  = (__bf16*)bump(p, (size_t)LL * CC * 2 * CC * 2);
    __bf16* x_b   = (__bf16*)bump(p, (size_t)NC * 2);
    __bf16* h_b   = (__bf16*)bump(p, (size_t)NC * 2);
    __bf16* out_bf= (__bf16*)bump(p, (size_t)NC * 2);
    __bf16* qkv_b = (__bf16*)bump(p, (size_t)NN * 384 * 2);
    __bf16* vt_b  = (__bf16*)bump(p, (size_t)NC * 2);
    __bf16* ao_b  = (__bf16*)bump(p, (size_t)NC * 2);
    __bf16* mh_b  = (__bf16*)bump(p, (size_t)NN * 256 * 2);
    float*  h     = (float*)bump(p, (size_t)NC * 4);
    float*  t0    = (float*)bump(p, (size_t)NC * 4);
    float*  loc   = (float*)bump(p, (size_t)NC * 4);
    float*  att   = (float*)bump(p, (size_t)NC * 4);
    float*  outf  = (float*)bump(p, (size_t)NC * 4);
    float*  t1    = (float*)bump(p, (size_t)NC * 4);
    float*  deg   = (float*)bump(p, (size_t)NN * 4);
    float*  dinv  = (float*)bump(p, (size_t)NN * 4);
    float*  logits= (float*)bump(p, (size_t)NN * NCLS * 4);
    float*  pooled= (float*)bump(p, (size_t)GG * NCLS * 4);
    (void)ws_size; (void)n_in; (void)out_size;

    // weight / input casts (recomputed every call: deterministic)
    cast_w(lin1_w, wl1_b, CC * CC, stream);
    cast_w(gcn_w,  wg_b,  LL * CC * CC, stream);
    cast_w(in_w,   wi_b,  LL * 3 * CC * CC, stream);
    cast_w(out_w,  wo_b,  LL * CC * CC, stream);
    cast_w(w1,     w1_b,  LL * 2 * CC * CC, stream);
    cast_w(w2,     w2_b,  LL * CC * 2 * CC, stream);
    cast_w(x,      x_b,   NC, stream);

    // h = relu(x @ lin1_w^T + lin1_b)
    launch_gemm(x_b, wl1_b, lin1_b, nullptr, h, h_b, NN, CC, CC, 1, stream);

    // degree (constant across layers): deg = 1 + in-degree
    fill_f32_k<<<NN / 256, 256, 0, stream>>>(deg, 1.0f, NN);
    deg_edge_k<<<(E + 255) / 256, 256, 0, stream>>>(dst, deg, E);
    dinv_k<<<NN / 256, 256, 0, stream>>>(deg, dinv, NN);

    for (int l = 0; l < LL; l++) {
        // ---- GCN branch: t0 = h @ gcn_w^T ; loc = scatter + self + h ; BN1
        launch_gemm(h_b, wg_b + (size_t)l * CC * CC, nullptr, nullptr,
                    t0, nullptr, NN, CC, CC, 0, stream);
        gcn_self_k<<<NC / 256, 256, 0, stream>>>(t0, dinv, h, loc, NC);
        gcn_scatter_k<<<(E * 32) / 256, 256, 0, stream>>>(t0, dinv, src, dst, loc, E);
        bn_k<<<CC, 256, 0, stream>>>(loc, n1g + l * CC, n1b + l * CC,
                                     loc, nullptr, NN, CC);

        // ---- Attention branch: qkv, V^T, flash, out-proj(+h), BN2
        launch_gemm(h_b, wi_b + (size_t)l * 3 * CC * CC, in_b + l * 3 * CC, nullptr,
                    nullptr, qkv_b, NN, 384, CC, 0, stream);
        vtrans_k<<<NC / 256, 256, 0, stream>>>(qkv_b, vt_b, NC);
        flash_attn_k<<<NN / 16, dim3(32, 4), 0, stream>>>(qkv_b, vt_b, ao_b);
        launch_gemm(ao_b, wo_b + (size_t)l * CC * CC, out_b + l * CC, h,
                    att, nullptr, NN, CC, CC, 0, stream);
        bn_k<<<CC, 256, 0, stream>>>(att, n2g + l * CC, n2b + l * CC,
                                     att, nullptr, NN, CC);

        // ---- combine + MLP(+residual) + BN3 -> h
        add2_k<<<NC / 256, 256, 0, stream>>>(loc, att, outf, out_bf, NC);
        launch_gemm(out_bf, w1_b + (size_t)l * 2 * CC * CC, b1 + l * 2 * CC, nullptr,
                    nullptr, mh_b, NN, 256, CC, 1, stream);
        launch_gemm(mh_b, w2_b + (size_t)l * CC * 2 * CC, b2 + l * CC, outf,
                    t1, nullptr, NN, CC, 2 * CC, 0, stream);
        bn_k<<<CC, 256, 0, stream>>>(t1, n3g + l * CC, n3b + l * CC,
                                     h, h_b, NN, CC);
    }

    // classifier head: logits -> segment-sum pool -> log_softmax
    lin2_k<<<(NN * NCLS + 255) / 256, 256, 0, stream>>>(h, lin2_w, lin2_b, logits);
    fill_f32_k<<<(GG * NCLS + 255) / 256, 256, 0, stream>>>(pooled, 0.0f, GG * NCLS);
    pool_k<<<(NN * NCLS + 255) / 256, 256, 0, stream>>>(logits, batch, pooled);
    lsm_k<<<1, 64, 0, stream>>>(pooled, (float*)d_out);
}
